// KSI_58712202936365
// MI455X (gfx1250) — matrix operand order repaired and checked by reading, not compiled, not run on hardware
//
#include <hip/hip_runtime.h>

// MI455X / gfx1250 fused kernel:
//   e    = (wikivec ⊙ notevec[n]) @ W_emb^T + b_emb   (bf16 WMMA, f32 acc, K=10000)
//   attn = sigmoid(e @ W_att^T + b_att)                (bf16 WMMA, f32 acc, K=256)
//   s    = sum_j attn*e*W_out + b_out                  (VALU + ds_add_f32)
// One workgroup = (n, 64-code block); e-tile lives entirely in LDS (bf16).
// Software-pipelined with peeled tail: steady-state loop has unguarded
// global_load_b128 prefetch; the single partial chunk pays guards once.

typedef __bf16 bf16;
typedef __attribute__((ext_vector_type(16))) __bf16 v16bf;
typedef __attribute__((ext_vector_type(8)))  __bf16 v8bf;
typedef __attribute__((ext_vector_type(8)))  float  v8f;
typedef __attribute__((ext_vector_type(4)))  float  v4f;

#define N_TOK   16
#define N_CODES 256
#define N_VOCAB 10000
#define N_EMB   256
#define CBLK    64      // codes per workgroup
#define KCH     32      // bf16 WMMA K per step

// A fragment (16x32 bf16): lane<16 -> row=lane, K 0..15; lane>=16 -> row=lane-16, K 16..31 (contiguous 32B)
__device__ __forceinline__ v16bf frag_a_lds(const bf16* rowp, int kb) {
  return *(const v16bf*)(rowp + 2 * kb);
}
// B fragment (32x16 bf16): lane holds col=lane&15, K = (lane>>4)*16 + 0..15 (contiguous 32B)
__device__ __forceinline__ v16bf frag_b_lds(const bf16* rowp, int lane) {
  return *(const v16bf*)(rowp + ((lane >> 4) << 4));
}
__device__ __forceinline__ v8f wmma_bf16(v16bf a, v16bf b, v8f c) {
  return __builtin_amdgcn_wmma_f32_16x16x32_bf16(false, a, false, b, (short)0, c, false, false);
}

// cooperative staging: A tile 64x32 (4 elems/thread), B tile 256x32 (16 elems/thread)
__device__ __forceinline__ void stage_ab(bf16* a_st, bf16* b_st,
                                         int arow, int acol, int brow, int bcol,
                                         const v4f& wv, const v4f& nv, const v4f (&bv)[4]) {
  #pragma unroll
  for (int i = 0; i < 4; ++i) a_st[arow*KCH + acol + i] = (bf16)(wv[i] * nv[i]);
  #pragma unroll
  for (int q = 0; q < 4; ++q)
    #pragma unroll
    for (int i = 0; i < 4; ++i) b_st[brow*KCH + bcol + q*4 + i] = (bf16)bv[q][i];
}

// 2x2 tile outer product: 8 ds_load_b128 -> 4 back-to-back WMMAs
__device__ __forceinline__ void mma_2x2(const bf16* a0, const bf16* a1,
                                        const bf16* b0, const bf16* b1,
                                        int kb, int lane, v8f (&acc)[2][2]) {
  v16bf af0 = frag_a_lds(a0, kb);
  v16bf af1 = frag_a_lds(a1, kb);
  v16bf bf0 = frag_b_lds(b0, lane);
  v16bf bf1 = frag_b_lds(b1, lane);
  acc[0][0] = wmma_bf16(af0, bf0, acc[0][0]);
  acc[0][1] = wmma_bf16(af0, bf1, acc[0][1]);
  acc[1][0] = wmma_bf16(af1, bf0, acc[1][0]);
  acc[1][1] = wmma_bf16(af1, bf1, acc[1][1]);
}

__global__ __launch_bounds__(512)
void KSI_58712202936365_kernel(const float* __restrict__ notevec,
                               const float* __restrict__ wikivec,
                               const float* __restrict__ W_emb,
                               const float* __restrict__ b_emb,
                               const float* __restrict__ W_att,
                               const float* __restrict__ b_att,
                               const float* __restrict__ W_out,
                               const float* __restrict__ b_out,
                               float* __restrict__ out) {
  __shared__ __align__(32) bf16 e_sh[CBLK * N_EMB];   // 32 KB: e tile (bf16)
  __shared__ __align__(32) bf16 a_st[CBLK * KCH];     //  4 KB: scaled wikivec chunk
  __shared__ __align__(32) bf16 b_st[N_EMB * KCH];    // 16 KB: W_emb / W_att chunk
  __shared__ float s_sh[CBLK];

  const int tid  = threadIdx.x;
  const int lane = tid & 31;
  const int w    = tid >> 5;            // wave 0..15
  const int ct0  = (w & 1) * 2;         // 2 c-subtiles per wave
  const int kt0  = (w >> 1) * 2;        // 2 k-subtiles per wave
  const int m    = lane & 15;
  const int kb   = (lane >> 4) << 3;    // 0 or 8; A-frag K base is 2*kb = 0 or 16
  const int hr   = (lane >> 4) << 3;    // C/D row offset: M = r + 8*(lane>=16)
  const int nb   = blockIdx.y;          // token n
  const int c0   = blockIdx.x * CBLK;   // code block base

  if (tid < CBLK) s_sh[tid] = 0.0f;

  const v4f z4 = {0,0,0,0};
  const v8f z8 = {0,0,0,0,0,0,0,0};
  v8f acc[2][2] = {{z8,z8},{z8,z8}};

  const int arow = tid >> 3, acol = (tid & 7) << 2;
  const int brow = tid >> 1, bcol = (tid & 1) << 4;
  const float* wrow   = wikivec + (size_t)(c0 + arow) * N_VOCAB;
  const float* nrow   = notevec + (size_t)nb * N_VOCAB;
  const float* brow_p = W_emb + (size_t)brow * N_VOCAB;

  // wave-private LDS bases for fragments
  const bf16* a_f0 = &a_st[((ct0    )*16 + m) * KCH];
  const bf16* a_f1 = &a_st[((ct0 + 1)*16 + m) * KCH];
  const bf16* b_f0 = &b_st[((kt0    )*16 + m) * KCH];
  const bf16* b_f1 = &b_st[((kt0 + 1)*16 + m) * KCH];

  // ---- Stage 1: e = (wikivec ⊙ notevec[n]) @ W_emb^T over K = 10000 ----
  const int FULLCH = N_VOCAB / KCH;        // 312 full chunks; one 16-wide tail
  const int TAIL_V0 = FULLCH * KCH;        // 9984
  const int REM = N_VOCAB - TAIL_V0;       // 16

  v4f wv, nv, bv[4], wv_n, nv_n, bv_n[4];
  // prologue: chunk 0 (full, unguarded)
  wv = *(const v4f*)(wrow + acol);
  nv = *(const v4f*)(nrow + acol);
  #pragma unroll
  for (int q = 0; q < 4; ++q) bv[q] = *(const v4f*)(brow_p + bcol + q*4);

  // steady state: chunks 0..FULLCH-2, prefetch ch+1 fully unguarded
  for (int ch = 0; ch < FULLCH - 1; ++ch) {
    stage_ab(a_st, b_st, arow, acol, brow, bcol, wv, nv, bv);
    __syncthreads();
    const int v0 = (ch + 1) * KCH;
    wv_n = *(const v4f*)(wrow + v0 + acol);
    nv_n = *(const v4f*)(nrow + v0 + acol);
    #pragma unroll
    for (int q = 0; q < 4; ++q) bv_n[q] = *(const v4f*)(brow_p + v0 + bcol + q*4);
    mma_2x2(a_f0, a_f1, b_f0, b_f1, kb, lane, acc);
    __syncthreads();
    wv = wv_n; nv = nv_n;
    #pragma unroll
    for (int q = 0; q < 4; ++q) bv[q] = bv_n[q];
  }

  // chunk FULLCH-1: compute full, prefetch the guarded 16-wide tail (guards run ONCE)
  {
    stage_ab(a_st, b_st, arow, acol, brow, bcol, wv, nv, bv);
    __syncthreads();
    wv_n = (acol + 4 <= REM) ? *(const v4f*)(wrow + TAIL_V0 + acol) : z4;
    nv_n = (acol + 4 <= REM) ? *(const v4f*)(nrow + TAIL_V0 + acol) : z4;
    #pragma unroll
    for (int q = 0; q < 4; ++q)
      bv_n[q] = (bcol + q*4 + 4 <= REM) ? *(const v4f*)(brow_p + TAIL_V0 + bcol + q*4) : z4;
    mma_2x2(a_f0, a_f1, b_f0, b_f1, kb, lane, acc);
    __syncthreads();
    wv = wv_n; nv = nv_n;
    #pragma unroll
    for (int q = 0; q < 4; ++q) bv[q] = bv_n[q];
  }

  // tail chunk: stage + compute, no prefetch
  {
    stage_ab(a_st, b_st, arow, acol, brow, bcol, wv, nv, bv);
    __syncthreads();
    mma_2x2(a_f0, a_f1, b_f0, b_f1, kb, lane, acc);
    __syncthreads();
  }

  // ---- e + b_emb -> LDS e-tile (bf16); D layout: M = r + 8*(lane>=16), N = lane&15 ----
  #pragma unroll
  for (int ci = 0; ci < 2; ++ci)
    #pragma unroll
    for (int ki = 0; ki < 2; ++ki) {
      const int kg = (kt0 + ki)*16 + m;
      const float bias = b_emb[kg];
      #pragma unroll
      for (int r = 0; r < 8; ++r)
        e_sh[((ct0 + ci)*16 + r + hr) * N_EMB + kg] = (bf16)(acc[ci][ki][r] + bias);
    }
  __syncthreads();

  // ---- Stage 2: logits = e @ W_att^T over K = 256 (8 chunks, pipelined) ----
  v8f att[2][2] = {{z8,z8},{z8,z8}};
  const float* arow_p = W_att + (size_t)brow * N_EMB;
  v4f cb[4], cb_n[4];
  #pragma unroll
  for (int q = 0; q < 4; ++q) cb[q] = *(const v4f*)(arow_p + bcol + q*4);

  for (int ch = 0; ch < N_EMB / KCH - 1; ++ch) {   // 0..6: unconditional prefetch
    #pragma unroll
    for (int q = 0; q < 4; ++q)
      #pragma unroll
      for (int i = 0; i < 4; ++i) b_st[brow*KCH + bcol + q*4 + i] = (bf16)cb[q][i];
    __syncthreads();
    const int k1 = (ch + 1) * KCH;
    #pragma unroll
    for (int q = 0; q < 4; ++q) cb_n[q] = *(const v4f*)(arow_p + k1 + bcol + q*4);
    const int k0 = ch * KCH;
    mma_2x2(&e_sh[((ct0)*16 + m) * N_EMB + k0], &e_sh[((ct0+1)*16 + m) * N_EMB + k0],
            b_f0, b_f1, kb, lane, att);
    __syncthreads();
    #pragma unroll
    for (int q = 0; q < 4; ++q) cb[q] = cb_n[q];
  }
  {   // last chunk, no prefetch, no trailing barrier needed
    #pragma unroll
    for (int q = 0; q < 4; ++q)
      #pragma unroll
      for (int i = 0; i < 4; ++i) b_st[brow*KCH + bcol + q*4 + i] = (bf16)cb[q][i];
    __syncthreads();
    const int k0 = N_EMB - KCH;
    mma_2x2(&e_sh[((ct0)*16 + m) * N_EMB + k0], &e_sh[((ct0+1)*16 + m) * N_EMB + k0],
            b_f0, b_f1, kb, lane, att);
  }

  // ---- sigmoid, v = attn*e, s = sum_j v * W_out[j] ----
  float part[2][8] = {{0,0,0,0,0,0,0,0},{0,0,0,0,0,0,0,0}};
  #pragma unroll
  for (int ci = 0; ci < 2; ++ci)
    #pragma unroll
    for (int ki = 0; ki < 2; ++ki) {
      const int jg = (kt0 + ki)*16 + m;
      const float ba = b_att[jg];
      const float wo = W_out[jg];
      #pragma unroll
      for (int r = 0; r < 8; ++r) {
        const int cl = (ct0 + ci)*16 + r + hr;
        const float logit = att[ci][ki][r] + ba;
        const float sg = 1.0f / (1.0f + __expf(-logit));   // v_exp_f32 path
        part[ci][r] += sg * (float)e_sh[cl * N_EMB + jg] * wo;
      }
    }
  #pragma unroll
  for (int ci = 0; ci < 2; ++ci)
    #pragma unroll
    for (int r = 0; r < 8; ++r)
      atomicAdd(&s_sh[(ct0 + ci)*16 + r + hr], part[ci][r]);   // ds_add_f32
  __syncthreads();
  if (tid < CBLK)
    out[(size_t)nb * N_CODES + c0 + tid] = s_sh[tid] + b_out[0];
}

extern "C" void kernel_launch(void* const* d_in, const int* in_sizes, int n_in,
                              void* d_out, int out_size, void* d_ws, size_t ws_size,
                              hipStream_t stream) {
  const float* notevec = (const float*)d_in[0];
  const float* wikivec = (const float*)d_in[1];
  const float* W_emb   = (const float*)d_in[2];
  const float* b_emb   = (const float*)d_in[3];
  const float* W_att   = (const float*)d_in[4];
  const float* b_att   = (const float*)d_in[5];
  const float* W_out   = (const float*)d_in[6];
  const float* b_out   = (const float*)d_in[7];
  float* out = (float*)d_out;

  dim3 grid(N_CODES / CBLK, N_TOK);  // (4, 16) workgroups, fully fused
  KSI_58712202936365_kernel<<<grid, 512, 0, stream>>>(
      notevec, wikivec, W_emb, b_emb, W_att, b_att, W_out, b_out, out);
}
